// MultislicePtychography_4758823763984
// MI455X (gfx1250) — matrix operand order repaired and checked by reading, not compiled, not run on hardware
//
#include <hip/hip_runtime.h>
#include <math.h>

// ---------------------------------------------------------------------------
// Multislice ptychography forward model on MI455X (gfx1250).
// fft2 is executed as two identical WMMA passes  L <- (L*M)^T  with the
// symmetric DFT matrix M (M*L*M == fft2(L)).  All LDS traffic is b128,
// all global traffic is global_load_b128/b64 from L2-resident tables.
// Wave state: 256x256 complex f16 in LDS (256 KB), one WG (16 wave32) per
// scan position, f32 accumulation in v_wmma_f32_16x16x32_f16.
// ---------------------------------------------------------------------------

typedef __attribute__((ext_vector_type(16))) _Float16 v16h;
typedef __attribute__((ext_vector_type(8)))  _Float16 h8;
typedef __attribute__((ext_vector_type(8)))  float    v8f;
typedef __attribute__((ext_vector_type(2)))  float    f2;   // POD complex

#define DD     256
#define NSLICE 8
#define OBJN   1024
#define PIF    3.14159265358979323846f
#define LAMF   0.0197f
#define PXF    0.1f
#define CONVF  0.025f
#define DZF    2.0f

// workspace byte offsets
#define WS_WF_RE 0u
#define WS_WF_IM (128u*1024u)
#define WS_WF_IN (256u*1024u)     // negated imag plane (for Cr accumulation)
#define WS_WI_RE (384u*1024u)
#define WS_WI_IM (512u*1024u)
#define WS_WI_IN (640u*1024u)
#define WS_H     (768u*1024u)     // f2[65536]
#define WS_PROBE (1280u*1024u)    // f2[65536]
#define WS_TMP   (1792u*1024u)    // f2[65536]
#define WS_NORM  (2304u*1024u)    // float

// explicit global-address-space access (forces global_load/store, not flat)
typedef const h8    __attribute__((address_space(1)))* gh8p;
typedef const float __attribute__((address_space(1)))* gf32p;
typedef const f2    __attribute__((address_space(1)))* gf2p;
typedef f2          __attribute__((address_space(1)))* gf2wp;
typedef const int   __attribute__((address_space(1)))* gi32p;

static __device__ __forceinline__ h8 gload8(const _Float16* p) {
  return *(gh8p)(uintptr_t)p;            // global_load_b128
}
static __device__ __forceinline__ v16h joinh(h8 lo, h8 hi) {
  return __builtin_shufflevector(lo, hi, 0,1,2,3,4,5,6,7,8,9,10,11,12,13,14,15);
}
static __device__ __forceinline__ v8f wmma16(v16h a, v16h b, v8f c) {
  return __builtin_amdgcn_wmma_f32_16x16x32_f16(false, a, false, b, (short)0, c,
                                                false, false);
}
static __device__ __forceinline__ float freq_of(int i) {
  return (float)(i < 128 ? i : i - 256) / (DD * PXF);
}

// ---------------------------------------------------------------------------
// Setup: DFT matrices (f16, + negated imag), propagator H, probe_k
// ---------------------------------------------------------------------------
__global__ void k_setup(const float* defocus, const float* Cs, const float* amag,
                        const float* aang, const float* asmth, char* ws) {
  int t = blockIdx.x * 256 + threadIdx.x;   // 0..65535
  int i = t >> 8, j = t & 255;

  _Float16* wfre = (_Float16*)(ws + WS_WF_RE);
  _Float16* wfim = (_Float16*)(ws + WS_WF_IM);
  _Float16* wfin = (_Float16*)(ws + WS_WF_IN);
  _Float16* wire = (_Float16*)(ws + WS_WI_RE);
  _Float16* wiim = (_Float16*)(ws + WS_WI_IM);
  _Float16* wiin = (_Float16*)(ws + WS_WI_IN);
  f2* H  = (f2*)(ws + WS_H);
  f2* pk = (f2*)(ws + WS_PROBE);

  // W[m,k] = exp(-2 pi i m k / 256) (symmetric);  Winv = conj(W)/256
  int mk = (i * j) & 255;
  float th = 2.0f * PIF * (float)mk / 256.0f;
  float sn, cs;  __sincosf(th, &sn, &cs);
  wfre[t] = (_Float16)cs;
  wfim[t] = (_Float16)(-sn);
  wfin[t] = (_Float16)sn;
  wire[t] = (_Float16)(cs * (1.0f / 256.0f));
  wiim[t] = (_Float16)(sn * (1.0f / 256.0f));
  wiin[t] = (_Float16)(-sn * (1.0f / 256.0f));

  float fx = freq_of(i), fy = freq_of(j);
  float k2 = fx * fx + fy * fy;
  float hs, hc;  __sincosf(PIF * LAMF * DZF * k2, &hs, &hc);
  H[t] = (f2){hc, hs};

  float kk = sqrtf(k2);
  float kang = atan2f(fy, fx);
  float df = defocus[0], c3 = Cs[0], am = amag[0], aa = aang[0], sm = asmth[0];
  float chi = PIF * LAMF * df * k2
            + 0.5f * PIF * LAMF * LAMF * LAMF * (c3 * 1.0e6f) * k2 * k2
            + PIF * LAMF * am * k2 * __cosf(2.0f * (kang - aa));
  float kmax = CONVF / LAMF;
  float smooth = fabsf(sm) + 0.01f;
  float z = (kmax - kk) / (kmax * smooth);
  float ap = 1.0f / (1.0f + __expf(-z));
  float ps, pc;  __sincosf(chi, &ps, &pc);
  pk[t] = (f2){ap * pc, ap * ps};

  if (t == 0) *(float*)(ws + WS_NORM) = 0.0f;
}

// probe ifft2, separable naive DFT (one-off, tiny): cols then rows
__global__ void k_probe_cols(char* ws) {
  int t = blockIdx.x * 256 + threadIdx.x;
  int m = t >> 8, v = t & 255;
  const f2* pk = (const f2*)(ws + WS_PROBE);
  f2* tmp = (f2*)(ws + WS_TMP);
  float ar = 0.f, ai = 0.f;
  for (int n = 0; n < DD; ++n) {
    float th = 2.0f * PIF * (float)((n * v) & 255) / 256.0f;
    float s, c;  __sincosf(th, &s, &c);
    f2 p = pk[m * DD + n];
    ar += p.x * c - p.y * s;
    ai += p.x * s + p.y * c;
  }
  tmp[m * DD + v] = (f2){ar, ai};
}

__global__ void k_probe_rows(char* ws) {
  __shared__ float red[256];
  int t = blockIdx.x * 256 + threadIdx.x;
  int u = t >> 8, v = t & 255;
  const f2* tmp = (const f2*)(ws + WS_TMP);
  f2* probe = (f2*)(ws + WS_PROBE);
  float ar = 0.f, ai = 0.f;
  for (int m = 0; m < DD; ++m) {
    float th = 2.0f * PIF * (float)((m * u) & 255) / 256.0f;
    float s, c;  __sincosf(th, &s, &c);
    f2 p = tmp[m * DD + v];
    ar += p.x * c - p.y * s;
    ai += p.x * s + p.y * c;
  }
  ar *= (1.0f / 65536.0f);
  ai *= (1.0f / 65536.0f);
  int us = (u + 128) & 255, vs = (v + 128) & 255;   // ifftshift
  probe[us * DD + vs] = (f2){ar, ai};
  red[threadIdx.x] = ar * ar + ai * ai;
  __syncthreads();
  for (int off = 128; off > 0; off >>= 1) {
    if (threadIdx.x < off) red[threadIdx.x] += red[threadIdx.x + off];
    __syncthreads();
  }
  if (threadIdx.x == 0) atomicAdd((float*)(ws + WS_NORM), red[0]);
}

__global__ void k_probe_norm(char* ws) {
  int t = blockIdx.x * 256 + threadIdx.x;
  f2* probe = (f2*)(ws + WS_PROBE);
  float sc = rsqrtf(*(const float*)(ws + WS_NORM));
  f2 p = probe[t];
  probe[t] = (f2){p.x * sc, p.y * sc};
}

// ---------------------------------------------------------------------------
// One WMMA pass:  L <- (L * M)^T   (M symmetric; two passes == M*L*M == fft2)
// 16 waves, wave w owns row block r0 = w*16.
//   A-frags: LDS rows (ds_load_b128), preloaded before barrier (WAR safety)
//   B-frags: rows of symmetric M from global (global_load_b128, L2-resident)
//   C store: transposed -> contiguous 8xf16 per lane (ds_store_b128)
//   Cr = Ar*Br + Ai*(-Bi) via pre-negated plane: zero VALU negation.
// ---------------------------------------------------------------------------
static __device__ void dft_pass(const _Float16* Mre, const _Float16* Mim,
                                const _Float16* MimN,
                                _Float16* XR, _Float16* XI) {
  const int lane = threadIdx.x & 31;
  const int w    = threadIdx.x >> 5;          // 0..15
  const int r0   = w * 16;
  const int n    = lane & 15;                 // B/C column within tile
  const int kb   = (lane < 16) ? 0 : 16;      // B K-offset per lane half
  const int moff = (lane < 16) ? 0 : 8;       // C row offset per lane half
  const int am   = lane & 15;                 // A row
  const int akb  = (lane < 16) ? 0 : 16;      // A K sub-block per lane half

  // Phase A: preload this wave's 16 rows as A-fragments (re & im)
  v16h Ar[8], Ai[8];
#pragma unroll
  for (int kc = 0; kc < 8; ++kc) {
    const _Float16* pr = XR + (r0 + am) * DD + kc * 32;
    const _Float16* pi = XI + (r0 + am) * DD + kc * 32;
    Ar[kc] = joinh(*(const h8*)(pr + akb), *(const h8*)(pr + 8 + akb));
    Ai[kc] = joinh(*(const h8*)(pi + akb), *(const h8*)(pi + 8 + akb));
  }
  __syncthreads();   // all A-frags in VGPRs before anyone overwrites LDS

  for (int ct = 0; ct < 16; ++ct) {
    const int c0 = ct * 16;
    const _Float16* mr   = Mre  + (c0 + n) * DD;   // symmetric: row == column
    const _Float16* mi   = Mim  + (c0 + n) * DD;
    const _Float16* min_ = MimN + (c0 + n) * DD;
    v8f cr = {}; v8f ci = {};
#pragma unroll
    for (int kc = 0; kc < 8; ++kc) {
      const int off = kc * 32 + kb;
      v16h Br  = joinh(gload8(mr   + off), gload8(mr   + off + 8));
      v16h Bi  = joinh(gload8(mi   + off), gload8(mi   + off + 8));
      v16h BiN = joinh(gload8(min_ + off), gload8(min_ + off + 8));
      cr = wmma16(Ar[kc], Br,  cr);     // + Ar*Br
      cr = wmma16(Ai[kc], BiN, cr);     // - Ai*Bi
      ci = wmma16(Ar[kc], Bi,  ci);     // + Ar*Bi
      ci = wmma16(Ai[kc], Br,  ci);     // + Ai*Br
    }
    if (ct + 1 < 16) {                  // stream next M rows through L2
      __builtin_prefetch(Mre + (c0 + 16 + n) * DD, 0, 1);
      __builtin_prefetch(Mim + (c0 + 16 + n) * DD, 0, 1);
    }
    h8 sr, si;
#pragma unroll
    for (int v = 0; v < 8; ++v) {
      sr[v] = (_Float16)cr[v];
      si[v] = (_Float16)ci[v];
    }
    // transposed store: array[(c0+n)][r0+moff .. +7]  (16B aligned)
    *(h8*)(XR + (c0 + n) * DD + r0 + moff) = sr;
    *(h8*)(XI + (c0 + n) * DD + r0 + moff) = si;
  }
}

// ---------------------------------------------------------------------------
// Main multislice kernel: one workgroup (512 thr / 16 wave32) per scan pos.
// ---------------------------------------------------------------------------
__global__ void __launch_bounds__(512)
k_multislice(const float* __restrict__ amp, const float* __restrict__ phs,
             const int* __restrict__ scan, const int* __restrict__ idxs,
             const char* __restrict__ ws, float* __restrict__ out) {
  extern __shared__ _Float16 smem[];          // 256 KB dynamic LDS
  _Float16* XR = smem;
  _Float16* XI = smem + DD * DD;

  const int b  = blockIdx.x;
  const int id = ((gi32p)(uintptr_t)idxs)[b];
  const int py = ((gi32p)(uintptr_t)scan)[id * 2 + 0];
  const int px = ((gi32p)(uintptr_t)scan)[id * 2 + 1];

  const _Float16* WfR = (const _Float16*)(ws + WS_WF_RE);
  const _Float16* WfI = (const _Float16*)(ws + WS_WF_IM);
  const _Float16* WfN = (const _Float16*)(ws + WS_WF_IN);
  const _Float16* WiR = (const _Float16*)(ws + WS_WI_RE);
  const _Float16* WiI = (const _Float16*)(ws + WS_WI_IM);
  const _Float16* WiN = (const _Float16*)(ws + WS_WI_IN);
  gf2p Hk  = (gf2p)(uintptr_t)(ws + WS_H);
  gf2p prb = (gf2p)(uintptr_t)(ws + WS_PROBE);
  gf32p ampg = (gf32p)(uintptr_t)amp;
  gf32p phsg = (gf32p)(uintptr_t)phs;

  // wave <- probe
  for (int t = threadIdx.x; t < DD * DD; t += 512) {
    f2 p = prb[t];
    XR[t] = (_Float16)p.x;
    XI[t] = (_Float16)p.y;
  }
  __syncthreads();

  for (int s = 0; s < NSLICE; ++s) {
    // wave *= amp * exp(i*phase)  (object patch gather; object is L2-resident)
    for (int t = threadIdx.x; t < DD * DD; t += 512) {
      const int r = t >> 8, c = t & 255;
      const int g = s * OBJN * OBJN + (py + r) * OBJN + (px + c);
      float a = ampg[g];
      float sn, cs;  __sincosf(phsg[g], &sn, &cs);
      float tr = a * cs, ti = a * sn;
      float wr = (float)XR[t], wi = (float)XI[t];
      XR[t] = (_Float16)(wr * tr - wi * ti);
      XI[t] = (_Float16)(wr * ti + wi * tr);
    }
    __syncthreads();
    // fft2
    dft_pass(WfR, WfI, WfN, XR, XI);  __syncthreads();
    dft_pass(WfR, WfI, WfN, XR, XI);  __syncthreads();
    // k-space propagator (H symmetric -> orientation-safe)
    for (int t = threadIdx.x; t < DD * DD; t += 512) {
      f2 h = Hk[t];
      float wr = (float)XR[t], wi = (float)XI[t];
      XR[t] = (_Float16)(wr * h.x - wi * h.y);
      XI[t] = (_Float16)(wr * h.y + wi * h.x);
    }
    __syncthreads();
    // ifft2  (Wi = conj(W)/256)
    dft_pass(WiR, WiI, WiN, XR, XI);  __syncthreads();
    dft_pass(WiR, WiI, WiN, XR, XI);  __syncthreads();
  }

  // psi = fft2(wave)
  dft_pass(WfR, WfI, WfN, XR, XI);  __syncthreads();
  dft_pass(WfR, WfI, WfN, XR, XI);  __syncthreads();

  gf2wp o = (gf2wp)(uintptr_t)(out + (size_t)b * DD * DD * 2);
  for (int t = threadIdx.x; t < DD * DD; t += 512) {
    o[t] = (f2){(float)XR[t], (float)XI[t]};   // global_store_b64
  }
}

// ---------------------------------------------------------------------------
extern "C" void kernel_launch(void* const* d_in, const int* in_sizes, int n_in,
                              void* d_out, int out_size, void* d_ws, size_t ws_size,
                              hipStream_t stream) {
  (void)in_sizes; (void)n_in; (void)out_size; (void)ws_size;
  const float* amp     = (const float*)d_in[0];
  const float* phs     = (const float*)d_in[1];
  const float* defocus = (const float*)d_in[2];
  const float* Cs      = (const float*)d_in[3];
  const float* amag    = (const float*)d_in[4];
  const float* aang    = (const float*)d_in[5];
  const float* asmth   = (const float*)d_in[6];
  const int*   scan    = (const int*)d_in[7];
  const int*   idxs    = (const int*)d_in[8];
  char*  ws  = (char*)d_ws;
  float* out = (float*)d_out;

  k_setup<<<256, 256, 0, stream>>>(defocus, Cs, amag, aang, asmth, ws);
  k_probe_cols<<<256, 256, 0, stream>>>(ws);
  k_probe_rows<<<256, 256, 0, stream>>>(ws);
  k_probe_norm<<<256, 256, 0, stream>>>(ws);

  (void)hipFuncSetAttribute((const void*)k_multislice,
                            hipFuncAttributeMaxDynamicSharedMemorySize,
                            256 * 1024);
  k_multislice<<<256, 512, 256 * 1024, stream>>>(amp, phs, scan, idxs, ws, out);
}
